// KeyJointACTGenerator_36515811950795
// MI455X (gfx1250) — matrix operand order repaired and checked by reading, not compiled, not run on hardware
//
#include <hip/hip_runtime.h>
#include <math.h>

// ---------------- CDNA5 (gfx1250) types ----------------
typedef __attribute__((ext_vector_type(16))) __bf16 v16bf;
typedef __attribute__((ext_vector_type(8)))  float  v8f;

// ---------------- problem constants ----------------
#define STATE_DIM 64
#define ACTION_DIM 32
#define NUM_INSTR 16
#define HIDDEN 1024
#define TRAJ 128
#define KJ 8
#define EMB 64
#define D_TEMP 1344          // HIDDEN + EMB + HIDDEN/4
#define NHEAD 8
#define DH 168               // D_TEMP / NHEAD
#define DFF 1024
#define NLAYERS 3
#define BATCH 256
#define LN_EPS 1e-5f
#define ROWS (BATCH * TRAJ)  // 32768

// ---------------- GEMM tiling ----------------
#define BM 128
#define BN 64
#define BK 32

__device__ __forceinline__ unsigned short f2bf(float f) {
  unsigned u = __float_as_uint(f);
  u += 0x7FFFu + ((u >> 16) & 1u);
  return (unsigned short)(u >> 16);
}

__device__ __forceinline__ float apply_act(float v, int act) {
  if (act == 1) return v > 0.f ? v : 0.f;           // relu
  if (act == 2) return 1.f / (1.f + __expf(-v));    // sigmoid
  return v;
}

// CDNA5 async global->LDS (VGLOBAL encoding, ASYNCcnt-tracked). Per-lane LDS
// destination VGPR lets us scatter tiles directly into the WMMA fragment
// layout with no VGPR bounce and no VALU conversion work.
__device__ __forceinline__ void async_b128_to_lds(unsigned lds_off, const void* gaddr) {
  asm volatile("global_load_async_to_lds_b128 %0, %1, off"
               :: "v"(lds_off), "v"(gaddr) : "memory");
}
__device__ __forceinline__ void wait_asynccnt0() {
  asm volatile("s_wait_asynccnt 0x0" ::: "memory");
}

// f32 -> bf16 conversion (weights once per launch; activations before each
// fast GEMM). bf16 operands then stream from the 192MB L2.
__global__ __launch_bounds__(256) void cvt_bf16_kernel(const float* __restrict__ src,
                                                       unsigned short* __restrict__ dst,
                                                       long n) {
  long i = (long)blockIdx.x * blockDim.x + threadIdx.x;
  long stride = (long)gridDim.x * blockDim.x;
  for (; i < n; i += stride) dst[i] = f2bf(src[i]);
}

// ---------------------------------------------------------------------------
// FAST GEMM: C[z][M,N] = act(A[z][M,K](bf16) * W[N,K](bf16)^T + bias[N])
// Requires M%128==0, N%64==0, K%32==0 (all heavy GEMMs in this net qualify).
// Both tiles staged via global_load_async_to_lds_b128 straight into the
// CDNA5 wave32 fragment layout; ping-pong LDS, one barrier per K-step.
// ---------------------------------------------------------------------------
__global__ __launch_bounds__(256) void gemm_wmma_fast(
    const unsigned short* __restrict__ A, const unsigned short* __restrict__ W,
    const float* __restrict__ bias, float* __restrict__ C,
    int M, int N, int K,
    long aBatch, long cBatch,
    const int* __restrict__ inst, long wExpert, int bExpert, int act) {
  __shared__ __align__(32) unsigned short sA[2][BM * BK]; // 2 x 8 KB
  __shared__ __align__(32) unsigned short sB[2][BN * BK]; // 2 x 4 KB

  const int z = blockIdx.z;
  const unsigned short* Az = A + (long)z * aBatch;
  float* Cz = C + (long)z * cBatch;
  const unsigned short* Wz = W;
  const float* bz = bias;
  if (inst) {
    int e = inst[z];
    e = e < 0 ? 0 : (e > NUM_INSTR - 1 ? NUM_INSTR - 1 : e);
    Wz += (long)e * wExpert;
    bz += (long)e * bExpert;
  }

  const int rowBase = blockIdx.y * BM;
  const int colBase = blockIdx.x * BN;
  const int tid  = threadIdx.x;
  const int lane = tid & 31;
  const int wid  = tid >> 5;
  const int wm   = wid & 3;      // wave rows: 32*wm
  const int wn   = wid >> 2;     // wave cols: 32*wn

  v8f acc[2][2] = {};

  const int kTiles = K / BK;
  for (int kt = 0; kt < kTiles; ++kt) {
    const int k0 = kt * BK;
    const int p  = kt & 1;
    const unsigned aBase = (unsigned)(size_t)(&sA[p][0]);
    const unsigned bBase = (unsigned)(size_t)(&sB[p][0]);

    // A tile 128x32 bf16: 512 16-byte chunks, 2 per thread.
    // A frag: lane = (m%16)+16*lg, halves h0-7 <-> k = 8*lg, h8-15 <-> k = 8*lg+16.
    #pragma unroll
    for (int i = 0; i < 2; ++i) {
      const int task = tid + i * 256;
      const int m = task >> 2, lg = task & 1, c = (task >> 1) & 1;
      const unsigned short* gsrc = Az + (long)(rowBase + m) * K + k0 + 8 * (lg + 2 * c);
      async_b128_to_lds(aBase + (unsigned)((m >> 4) * 1024 + ((m & 15) + (lg << 4)) * 32 + (c << 4)),
                        gsrc);
    }
    // B tile 64(n) x 32(k) bf16: 256 16-byte chunks, same fragment layout as A
    // (lane = (n%16)+16*lg, h0-7 <-> k = 8*lg, h8-15 <-> k = 8*lg+16).
    {
      const int n = tid >> 2, lg = tid & 1, c = (tid >> 1) & 1;
      const unsigned short* gsrc = Wz + (long)(colBase + n) * K + k0 + 8 * (lg + 2 * c);
      async_b128_to_lds(bBase + (unsigned)((n >> 4) * 1024 + ((n & 15) + (lg << 4)) * 32 + (c << 4)),
                        gsrc);
    }
    wait_asynccnt0();
    __syncthreads(); // single barrier per K-step (ping-pong buffers)

    const v16bf a0 = *(const v16bf*)(sA[p] + (2 * wm + 0) * 512 + lane * 16);
    const v16bf a1 = *(const v16bf*)(sA[p] + (2 * wm + 1) * 512 + lane * 16);
    const v16bf b0 = *(const v16bf*)(sB[p] + (2 * wn + 0) * 512 + lane * 16);
    const v16bf b1 = *(const v16bf*)(sB[p] + (2 * wn + 1) * 512 + lane * 16);
    acc[0][0] = __builtin_amdgcn_wmma_f32_16x16x32_bf16(false, a0, false, b0, (short)0, acc[0][0], false, false);
    acc[0][1] = __builtin_amdgcn_wmma_f32_16x16x32_bf16(false, a0, false, b1, (short)0, acc[0][1], false, false);
    acc[1][0] = __builtin_amdgcn_wmma_f32_16x16x32_bf16(false, a1, false, b0, (short)0, acc[1][0], false, false);
    acc[1][1] = __builtin_amdgcn_wmma_f32_16x16x32_bf16(false, a1, false, b1, (short)0, acc[1][1], false, false);
  }

  #pragma unroll
  for (int i = 0; i < 2; ++i)
    #pragma unroll
    for (int j = 0; j < 2; ++j) {
      const int col = colBase + 32 * wn + 16 * j + (lane & 15);
      const int rb  = rowBase + 32 * wm + 16 * i + ((lane >> 4) << 3);
      const float bv = bz[col];
      #pragma unroll
      for (int r = 0; r < 8; ++r)
        Cz[(long)(rb + r) * N + col] = apply_act(acc[i][j][r] + bv, act);
    }
}

// ---------------------------------------------------------------------------
// GENERAL GEMM (guarded, f32 A): only the tiny ragged MLPs land here.
// ---------------------------------------------------------------------------
__global__ __launch_bounds__(256) void gemm_wmma_gen(
    const float* __restrict__ A, const unsigned short* __restrict__ W,
    const float* __restrict__ bias, float* __restrict__ C,
    int M, int N, int K,
    long aBatch, long cBatch,
    const int* __restrict__ inst, long wExpert, int bExpert, int act) {
  __shared__ __align__(32) unsigned short sA[2][BM * BK];
  __shared__ __align__(32) unsigned short sB[2][BN * BK];

  const int z = blockIdx.z;
  const float* Az = A + (long)z * aBatch;
  float* Cz = C + (long)z * cBatch;
  const unsigned short* Wz = W;
  const float* bz = bias;
  if (inst) {
    int e = inst[z];
    e = e < 0 ? 0 : (e > NUM_INSTR - 1 ? NUM_INSTR - 1 : e);
    Wz += (long)e * wExpert;
    bz += (long)e * bExpert;
  }

  const int rowBase = blockIdx.y * BM;
  const int colBase = blockIdx.x * BN;
  const int tid  = threadIdx.x;
  const int lane = tid & 31;
  const int wid  = tid >> 5;
  const int wm   = wid & 3;
  const int wn   = wid >> 2;

  v8f acc[2][2] = {};

  const int kTiles = (K + BK - 1) / BK;
  for (int kt = 0; kt < kTiles; ++kt) {
    const int k0 = kt * BK;
    const int p  = kt & 1;
    #pragma clang loop unroll(disable)
    for (int i = 0; i < 16; ++i) {
      const int idx = tid + i * 256;
      const int m = idx >> 5, kk = idx & 31;
      const int gr = rowBase + m, gk = k0 + kk;
      float v = 0.f;
      if (gr < M && gk < K) v = Az[(long)gr * K + gk];
      sA[p][(m >> 4) * 512 + ((m & 15) + (((kk >> 3) & 1) << 4)) * 16 +
            ((kk & 7) + ((kk >> 4) << 3))] = f2bf(v);
    }
    #pragma clang loop unroll(disable)
    for (int i = 0; i < 8; ++i) {
      const int idx = tid + i * 256;
      const int n = idx >> 5, kk = idx & 31;
      const int gn = colBase + n, gk = k0 + kk;
      unsigned short v = 0;
      if (gn < N && gk < K) v = Wz[(long)gn * K + gk];
      sB[p][(n >> 4) * 512 + ((n & 15) + (((kk >> 3) & 1) << 4)) * 16 +
            ((kk & 7) + ((kk >> 4) << 3))] = v;
    }
    __syncthreads();

    const v16bf a0 = *(const v16bf*)(sA[p] + (2 * wm + 0) * 512 + lane * 16);
    const v16bf a1 = *(const v16bf*)(sA[p] + (2 * wm + 1) * 512 + lane * 16);
    const v16bf b0 = *(const v16bf*)(sB[p] + (2 * wn + 0) * 512 + lane * 16);
    const v16bf b1 = *(const v16bf*)(sB[p] + (2 * wn + 1) * 512 + lane * 16);
    acc[0][0] = __builtin_amdgcn_wmma_f32_16x16x32_bf16(false, a0, false, b0, (short)0, acc[0][0], false, false);
    acc[0][1] = __builtin_amdgcn_wmma_f32_16x16x32_bf16(false, a0, false, b1, (short)0, acc[0][1], false, false);
    acc[1][0] = __builtin_amdgcn_wmma_f32_16x16x32_bf16(false, a1, false, b0, (short)0, acc[1][0], false, false);
    acc[1][1] = __builtin_amdgcn_wmma_f32_16x16x32_bf16(false, a1, false, b1, (short)0, acc[1][1], false, false);
  }

  #pragma unroll
  for (int i = 0; i < 2; ++i)
    #pragma unroll
    for (int j = 0; j < 2; ++j) {
      const int col = colBase + 32 * wn + 16 * j + (lane & 15);
      const int rb  = rowBase + 32 * wm + 16 * i + ((lane >> 4) << 3);
      #pragma clang loop unroll(disable)
      for (int r = 0; r < 8; ++r) {
        const int row = rb + r;
        if (row < M && col < N)
          Cz[(long)row * N + col] = apply_act(acc[i][j][r] + bz[col], act);
      }
    }
}

// ---------------- attention: one block per (head, batch), thread = query ----------------
__global__ __launch_bounds__(128) void attn_kernel(const float* __restrict__ qkv,
                                                   float* __restrict__ ctx) {
  const int h = blockIdx.x, b = blockIdx.y, q = threadIdx.x;
  __shared__ float sc[TRAJ * TRAJ]; // 64 KB of 320 KB WGP LDS
  float* my = sc + q * TRAJ;
  const long rs = 3L * D_TEMP;
  const float* qrow = qkv + ((long)b * TRAJ + q) * rs + h * DH;
  const float scale = rsqrtf((float)DH);
  for (int k = 0; k < TRAJ; ++k) {
    const float* krow = qkv + ((long)b * TRAJ + k) * rs + D_TEMP + h * DH;
    float acc = 0.f;
    for (int d = 0; d < DH; ++d) acc += qrow[d] * krow[d];
    my[k] = acc * scale;
  }
  float mx = my[0];
  for (int k = 1; k < TRAJ; ++k) mx = my[k] > mx ? my[k] : mx;
  float s = 0.f;
  for (int k = 0; k < TRAJ; ++k) { float e = __expf(my[k] - mx); my[k] = e; s += e; }
  const float invs = 1.f / s;
  float* crow = ctx + ((long)b * TRAJ + q) * D_TEMP + h * DH;
  for (int d = 0; d < DH; ++d) {
    float acc = 0.f;
    for (int k = 0; k < TRAJ; ++k)
      acc += my[k] * qkv[((long)b * TRAJ + k) * rs + 2 * D_TEMP + h * DH + d];
    crow[d] = acc * invs;
  }
}

// ---------------- residual + LayerNorm (in place on x) ----------------
__global__ __launch_bounds__(256) void resid_ln_kernel(float* __restrict__ x,
    const float* __restrict__ r, const float* __restrict__ g,
    const float* __restrict__ b, int D) {
  const long row = blockIdx.x;
  float* xr = x + row * D;
  const float* rr = r + row * D;
  __shared__ float s1[256], s2[256];
  float sum = 0.f, sq = 0.f;
  for (int i = threadIdx.x; i < D; i += 256) {
    float t = xr[i] + rr[i];
    sum += t; sq += t * t;
  }
  s1[threadIdx.x] = sum; s2[threadIdx.x] = sq;
  __syncthreads();
  for (int off = 128; off > 0; off >>= 1) {
    if (threadIdx.x < off) {
      s1[threadIdx.x] += s1[threadIdx.x + off];
      s2[threadIdx.x] += s2[threadIdx.x + off];
    }
    __syncthreads();
  }
  const float mean = s1[0] / D;
  const float var  = s2[0] / D - mean * mean;
  const float inv  = rsqrtf(var + LN_EPS);
  for (int i = threadIdx.x; i < D; i += 256) {
    float t = xr[i] + rr[i];
    xr[i] = (t - mean) * inv * g[i] + b[i];
  }
}

// ---------------- small builders ----------------
__global__ void fill_t_kernel(float* __restrict__ t) {
  int i = threadIdx.x;
  if (i < TRAJ) t[i] = (float)i / (float)(TRAJ - 1);
}

__global__ __launch_bounds__(256) void build_impin_kernel(const float* __restrict__ se,
    const float* __restrict__ emb, const int* __restrict__ inst, float* __restrict__ out) {
  const int b = blockIdx.x;
  int e = inst[b]; e = e < 0 ? 0 : (e > NUM_INSTR - 1 ? NUM_INSTR - 1 : e);
  float* o = out + (long)b * (HIDDEN + EMB);
  for (int i = threadIdx.x; i < HIDDEN + EMB; i += 256)
    o[i] = (i < HIDDEN) ? se[(long)b * HIDDEN + i] : emb[(long)e * EMB + (i - HIDDEN)];
}

__global__ __launch_bounds__(256) void build_x_kernel(const float* __restrict__ se,
    const float* __restrict__ emb, const int* __restrict__ inst,
    const float* __restrict__ te, float* __restrict__ x) {
  const long row = blockIdx.x;
  const int b = (int)(row / TRAJ), t = (int)(row % TRAJ);
  int e = inst[b]; e = e < 0 ? 0 : (e > NUM_INSTR - 1 ? NUM_INSTR - 1 : e);
  float* xr = x + row * D_TEMP;
  for (int i = threadIdx.x; i < D_TEMP; i += 256) {
    float v;
    if (i < HIDDEN)            v = se[(long)b * HIDDEN + i];
    else if (i < HIDDEN + EMB) v = emb[(long)e * EMB + (i - HIDDEN)];
    else                       v = te[(long)t * (HIDDEN / 4) + (i - HIDDEN - EMB)];
    xr[i] = v;
  }
}

// ---------------- host orchestration ----------------
extern "C" void kernel_launch(void* const* d_in, const int* in_sizes, int n_in,
                              void* d_out, int out_size, void* d_ws, size_t ws_size,
                              hipStream_t stream) {
  (void)in_sizes; (void)n_in; (void)out_size; (void)ws_size;
  const float* start = (const float*)d_in[0];
  const int*   inst  = (const int*)d_in[1];
  const float* cls0_b = (const float*)d_in[2],  *cls0_w = (const float*)d_in[3];
  const float* cls1_b = (const float*)d_in[4],  *cls1_w = (const float*)d_in[5];
  const float* cls2_b = (const float*)d_in[6],  *cls2_w = (const float*)d_in[7];
  const float* emb    = (const float*)d_in[8];
  const float* exp0_b = (const float*)d_in[9],  *exp0_w = (const float*)d_in[10];
  const float* exp1_b = (const float*)d_in[11], *exp1_w = (const float*)d_in[12];
  const float* ex_b1 = (const float*)d_in[13], *ex_b2 = (const float*)d_in[14], *ex_b3 = (const float*)d_in[15];
  const float* ex_w1 = (const float*)d_in[16], *ex_w2 = (const float*)d_in[17], *ex_w3 = (const float*)d_in[18];
  const float* imp0_b = (const float*)d_in[19], *imp0_w = (const float*)d_in[20];
  const float* imp1_b = (const float*)d_in[21], *imp1_w = (const float*)d_in[22];
  const float *L_ff1b[NLAYERS], *L_ff1w[NLAYERS], *L_ff2b[NLAYERS], *L_ff2w[NLAYERS];
  const float *L_inb[NLAYERS], *L_inw[NLAYERS];
  const float *L_ln1b[NLAYERS], *L_ln1g[NLAYERS], *L_ln2b[NLAYERS], *L_ln2g[NLAYERS];
  const float *L_outb[NLAYERS], *L_outw[NLAYERS];
  for (int l = 0; l < NLAYERS; ++l) {
    const int base = 23 + 12 * l;
    L_ff1b[l] = (const float*)d_in[base + 0];  L_ff1w[l] = (const float*)d_in[base + 1];
    L_ff2b[l] = (const float*)d_in[base + 2];  L_ff2w[l] = (const float*)d_in[base + 3];
    L_inb[l]  = (const float*)d_in[base + 4];  L_inw[l]  = (const float*)d_in[base + 5];
    L_ln1b[l] = (const float*)d_in[base + 6];  L_ln1g[l] = (const float*)d_in[base + 7];
    L_ln2b[l] = (const float*)d_in[base + 8];  L_ln2g[l] = (const float*)d_in[base + 9];
    L_outb[l] = (const float*)d_in[base + 10]; L_outw[l] = (const float*)d_in[base + 11];
  }
  const float* se0_b = (const float*)d_in[59], *se0_w = (const float*)d_in[60];
  const float* se1_b = (const float*)d_in[61], *se1_w = (const float*)d_in[62];
  const float* t0_b  = (const float*)d_in[63], *t0_w  = (const float*)d_in[64];
  const float* t1_b  = (const float*)d_in[65], *t1_w  = (const float*)d_in[66];

  float* out = (float*)d_out;
  float* out_full = out;                                   // [256,128,32]
  float* out_log  = out + (long)ROWS * ACTION_DIM;         // [256,16]
  float* out_imp  = out_log + BATCH * NUM_INSTR;           // [256,32]
  float* out_kja  = out_imp + BATCH * ACTION_DIM;          // [256,128,8]

  char* wsb = (char*)d_ws;
  size_t cur = 0;
  auto allocb = [&](size_t nbytes) -> char* {
    char* p = wsb + cur;
    cur += ((nbytes + 255) / 256) * 256;
    return p;
  };
  auto allocf = [&](size_t n) -> float* { return (float*)allocb(n * 4); };

  // ---- one-time f32 -> bf16 weight conversion ----
  auto cvt = [&](const float* src, size_t n) -> const unsigned short* {
    unsigned short* dst = (unsigned short*)allocb(n * 2);
    cvt_bf16_kernel<<<1024, 256, 0, stream>>>(src, dst, (long)n);
    return dst;
  };
  const unsigned short* b_se0 = cvt(se0_w, (size_t)HIDDEN * STATE_DIM);
  const unsigned short* b_se1 = cvt(se1_w, (size_t)HIDDEN * HIDDEN);
  const unsigned short* b_cls0 = cvt(cls0_w, (size_t)128 * HIDDEN);
  const unsigned short* b_cls1 = cvt(cls1_w, (size_t)64 * 128);
  const unsigned short* b_cls2 = cvt(cls2_w, (size_t)NUM_INSTR * 64);
  const unsigned short* b_imp0 = cvt(imp0_w, (size_t)128 * (HIDDEN + EMB));
  const unsigned short* b_imp1 = cvt(imp1_w, (size_t)ACTION_DIM * 128);
  const unsigned short* b_t0 = cvt(t0_w, (size_t)(HIDDEN / 4) * 1);
  const unsigned short* b_t1 = cvt(t1_w, (size_t)(HIDDEN / 4) * (HIDDEN / 4));
  const unsigned short* b_exp0 = cvt(exp0_w, (size_t)128 * KJ);
  const unsigned short* b_exp1 = cvt(exp1_w, (size_t)ACTION_DIM * 128);
  const unsigned short* b_w1 = cvt(ex_w1, (size_t)NUM_INSTR * 256 * D_TEMP);
  const unsigned short* b_w2 = cvt(ex_w2, (size_t)NUM_INSTR * 128 * 256);
  const unsigned short* b_w3 = cvt(ex_w3, (size_t)NUM_INSTR * KJ * 128);
  const unsigned short *b_inw[NLAYERS], *b_outw[NLAYERS], *b_ff1w[NLAYERS], *b_ff2w[NLAYERS];
  for (int l = 0; l < NLAYERS; ++l) {
    b_inw[l]  = cvt(L_inw[l],  (size_t)3 * D_TEMP * D_TEMP);
    b_outw[l] = cvt(L_outw[l], (size_t)D_TEMP * D_TEMP);
    b_ff1w[l] = cvt(L_ff1w[l], (size_t)DFF * D_TEMP);
    b_ff2w[l] = cvt(L_ff2w[l], (size_t)D_TEMP * DFF);
  }

  // ---- activation scratch ----
  unsigned short* abf = (unsigned short*)allocb((size_t)ROWS * D_TEMP * 2); // bf16 A staging
  float* seh   = allocf((size_t)BATCH * HIDDEN);
  float* se    = allocf((size_t)BATCH * HIDDEN);
  float* h128  = allocf((size_t)BATCH * 128);
  float* h64   = allocf((size_t)BATCH * 64);
  float* impin = allocf((size_t)BATCH * (HIDDEN + EMB));
  float* imph  = allocf((size_t)BATCH * 128);
  float* tbuf  = allocf(TRAJ);
  float* teh   = allocf((size_t)TRAJ * (HIDDEN / 4));
  float* te    = allocf((size_t)TRAJ * (HIDDEN / 4));
  float* x     = allocf((size_t)ROWS * D_TEMP);
  float* qkv   = allocf((size_t)ROWS * 3 * D_TEMP);
  float* ctx   = allocf((size_t)ROWS * D_TEMP);
  float* ffb   = allocf((size_t)ROWS * DFF);
  float* tmp   = qkv;                   // out-proj / ff2 output: qkv dead by then
  float* k1b   = ctx;                   // expert buffers: ctx dead after layers
  float* k2b   = ctx + (size_t)ROWS * 256;
  float* exph  = k2b + (size_t)ROWS * 128;

  // fast GEMM: converts A to bf16 first (same downconvert the kernel used to
  // do in-LDS), then runs the pure async+WMMA kernel.
  auto gemmF = [&](const float* A, const unsigned short* W, const float* bias, float* C,
                   int M, int N, int K, int act, int Z = 1, long aB = 0, long cB = 0,
                   const int* e = nullptr, long wS = 0, int bS = 0) {
    const long an = (Z - 1) * aB + (long)M * K;
    cvt_bf16_kernel<<<1024, 256, 0, stream>>>(A, abf, an);
    dim3 g(N / BN, M / BM, Z);
    gemm_wmma_fast<<<g, 256, 0, stream>>>(abf, W, bias, C, M, N, K, aB, cB, e, wS, bS, act);
  };
  auto gemmG = [&](const float* A, const unsigned short* W, const float* bias, float* C,
                   int M, int N, int K, int act, int Z = 1, long aB = 0, long cB = 0,
                   const int* e = nullptr, long wS = 0, int bS = 0) {
    dim3 g((N + BN - 1) / BN, (M + BM - 1) / BM, Z);
    gemm_wmma_gen<<<g, 256, 0, stream>>>(A, W, bias, C, M, N, K, aB, cB, e, wS, bS, act);
  };

  // ---- state encoder ----
  gemmF(start, b_se0, se0_b, seh, BATCH, HIDDEN, STATE_DIM, 1);
  gemmF(seh, b_se1, se1_b, se, BATCH, HIDDEN, HIDDEN, 0);
  // ---- instruction classifier -> logits ----
  gemmF(se, b_cls0, cls0_b, h128, BATCH, 128, HIDDEN, 1);
  gemmF(h128, b_cls1, cls1_b, h64, BATCH, 64, 128, 1);
  gemmG(h64, b_cls2, cls2_b, out_log, BATCH, NUM_INSTR, 64, 0);
  // ---- joint importance ----
  build_impin_kernel<<<BATCH, 256, 0, stream>>>(se, emb, inst, impin);
  gemmF(impin, b_imp0, imp0_b, imph, BATCH, 128, HIDDEN + EMB, 1);
  gemmG(imph, b_imp1, imp1_b, out_imp, BATCH, ACTION_DIM, 128, 2);
  // ---- time encoding ----
  fill_t_kernel<<<1, 128, 0, stream>>>(tbuf);
  gemmG(tbuf, b_t0, t0_b, teh, TRAJ, HIDDEN / 4, 1, 1);
  gemmF(teh, b_t1, t1_b, te, TRAJ, HIDDEN / 4, HIDDEN / 4, 0);
  // ---- temporal input ----
  build_x_kernel<<<ROWS, 256, 0, stream>>>(se, emb, inst, te, x);
  // ---- transformer layers ----
  for (int l = 0; l < NLAYERS; ++l) {
    gemmF(x, b_inw[l], L_inb[l], qkv, ROWS, 3 * D_TEMP, D_TEMP, 0);
    attn_kernel<<<dim3(NHEAD, BATCH), 128, 0, stream>>>(qkv, ctx);
    gemmF(ctx, b_outw[l], L_outb[l], tmp, ROWS, D_TEMP, D_TEMP, 0);
    resid_ln_kernel<<<ROWS, 256, 0, stream>>>(x, tmp, L_ln1g[l], L_ln1b[l], D_TEMP);
    gemmF(x, b_ff1w[l], L_ff1b[l], ffb, ROWS, DFF, D_TEMP, 1);
    gemmF(ffb, b_ff2w[l], L_ff2b[l], tmp, ROWS, D_TEMP, DFF, 0);
    resid_ln_kernel<<<ROWS, 256, 0, stream>>>(x, tmp, L_ln2g[l], L_ln2b[l], D_TEMP);
  }
  // ---- grouped expert MLP (per-sample weights via instruction id) ----
  gemmF(x, b_w1, ex_b1, k1b, TRAJ, 256, D_TEMP, 1, BATCH,
        (long)TRAJ * D_TEMP, (long)TRAJ * 256, inst, 256L * D_TEMP, 256);
  gemmF(k1b, b_w2, ex_b2, k2b, TRAJ, 128, 256, 1, BATCH,
        (long)TRAJ * 256, (long)TRAJ * 128, inst, 128L * 256, 128);
  gemmG(k2b, b_w3, ex_b3, out_kja, TRAJ, KJ, 128, 0, BATCH,
        (long)TRAJ * 128, (long)TRAJ * KJ, inst, (long)KJ * 128, KJ);
  // ---- full joint expander ----
  gemmG(out_kja, b_exp0, exp0_b, exph, ROWS, 128, KJ, 1);
  gemmG(exph, b_exp1, exp1_b, out_full, ROWS, ACTION_DIM, 128, 0);
}
